// Transcoder_39676907888275
// MI455X (gfx1250) — matrix operand order repaired and checked by reading, not compiled, not run on hardware
//
#include <hip/hip_runtime.h>
#include <hip/hip_bf16.h>
#include <cstdint>

typedef __attribute__((ext_vector_type(16))) __bf16 v16bf;
typedef __attribute__((ext_vector_type(8)))  float  v8f;

constexpr int D_IN  = 2048;
constexpr int D_OUT = 2048;
constexpr int D_HID = 16384;
constexpr int NTOK  = 8192;
constexpr int TOPK  = 64;

// Low 32 bits of a generic pointer into LDS == the LDS byte address (AS3),
// per CDNA5 flat addressing.
__device__ inline unsigned lds_addr(const void* p) {
  return (unsigned)(uintptr_t)p;
}

__device__ inline void pack8(v16bf& f, int base, float4 a, float4 b) {
  f[base+0]=(__bf16)a.x; f[base+1]=(__bf16)a.y; f[base+2]=(__bf16)a.z; f[base+3]=(__bf16)a.w;
  f[base+4]=(__bf16)b.x; f[base+5]=(__bf16)b.y; f[base+6]=(__bf16)b.z; f[base+7]=(__bf16)b.w;
}

// ---------------------------------------------------------------------------
// C[M,N] = A[M,K] @ B[N,K]^T + bias0[N] (+ bias1[N]); both operands K-major.
// Block tile 128x128 (8 waves, each a 32x64 sub-tile of 16x16x32 bf16 WMMAs).
// K streamed in 32-wide chunks, double-buffered in LDS via async loads:
//   issue chunk k+1 -> s_wait_asynccnt 8 (chunk k complete, in-order) ->
//   barrier -> WMMA on chunk k -> barrier.
// ---------------------------------------------------------------------------
constexpr int BM = 128, BN = 128, KC = 32;
constexpr int LSTRIDE = 36;  // floats per LDS row (128B data + 16B pad)

template<bool HAS_B1>
__global__ __launch_bounds__(256) void gemm_nt_bias(
    const float* __restrict__ A, const float* __restrict__ B,
    const float* __restrict__ bias0, const float* __restrict__ bias1,
    float* __restrict__ C, int M, int N, int K)
{
  __shared__ __align__(16) float Abuf[2][BM * LSTRIDE];  // 2 x 18 KB
  __shared__ __align__(16) float Bbuf[2][BN * LSTRIDE];  // 2 x 18 KB

  const int tid  = threadIdx.x;
  const int lane = tid & 31;
  const int wid  = tid >> 5;      // 0..7
  const int wm   = wid >> 1;      // wave row 0..3  (32 rows each)
  const int wn   = wid & 1;       // wave col 0..1  (64 cols each)
  const int half = lane >> 4;
  const int l16  = lane & 15;

  const int tilesN = N / BN;
  const int m0 = (blockIdx.x / tilesN) * BM;
  const int n0 = (blockIdx.x % tilesN) * BN;

  // 256 threads cooperatively copy one 128x32 f32 chunk of A and of B:
  // 1024 float4 each -> 4 async b128 per thread per matrix.
  auto issueChunk = [&](int k0, int buf) {
    const unsigned aBase = lds_addr(&Abuf[buf][0]);
    const unsigned bBase = lds_addr(&Bbuf[buf][0]);
#pragma unroll
    for (int i = 0; i < 4; ++i) {
      const int q   = tid + 256 * i;
      const int row = q >> 3;            // 0..127
      const int kq  = (q & 7) * 4;       // 0,4,..,28
      const unsigned ldst = (unsigned)((row * LSTRIDE + kq) * sizeof(float));
      const unsigned aOff = (unsigned)(((size_t)(m0 + row) * K + k0 + kq) * sizeof(float));
      asm volatile("global_load_async_to_lds_b128 %0, %1, %2"
                   :: "v"(aBase + ldst), "v"(aOff), "s"(A) : "memory");
      const unsigned bOff = (unsigned)(((size_t)(n0 + row) * K + k0 + kq) * sizeof(float));
      asm volatile("global_load_async_to_lds_b128 %0, %1, %2"
                   :: "v"(bBase + ldst), "v"(bOff), "s"(B) : "memory");
    }
  };

  v8f acc[2][4];
  const v8f vzero = {0.f,0.f,0.f,0.f,0.f,0.f,0.f,0.f};
#pragma unroll
  for (int mi = 0; mi < 2; ++mi)
#pragma unroll
    for (int ni = 0; ni < 4; ++ni) acc[mi][ni] = vzero;

  issueChunk(0, 0);

  const int nch = K / KC;
  for (int kc = 0; kc < nch; ++kc) {
    const int cur = kc & 1;
    if (kc + 1 < nch) {
      issueChunk((kc + 1) * KC, cur ^ 1);
      asm volatile("s_wait_asynccnt 0x8" ::: "memory");  // chunk kc done (in-order)
    } else {
      asm volatile("s_wait_asynccnt 0x0" ::: "memory");
    }
    __syncthreads();

    v16bf afrag[2];
#pragma unroll
    for (int mi = 0; mi < 2; ++mi) {
      const float* ap = &Abuf[cur][(wm*32 + mi*16 + l16) * LSTRIDE + half*8];
      pack8(afrag[mi], 0, *(const float4*)(ap +  0), *(const float4*)(ap +  4));
      pack8(afrag[mi], 8, *(const float4*)(ap + 16), *(const float4*)(ap + 20));
    }
    v16bf bfrag[4];
#pragma unroll
    for (int ni = 0; ni < 4; ++ni) {
      const float* bp = &Bbuf[cur][(wn*64 + ni*16 + l16) * LSTRIDE + half*8];
      pack8(bfrag[ni], 0, *(const float4*)(bp + 0), *(const float4*)(bp + 4));
      pack8(bfrag[ni], 8, *(const float4*)(bp + 16), *(const float4*)(bp + 20));
    }
#pragma unroll
    for (int mi = 0; mi < 2; ++mi)
#pragma unroll
      for (int ni = 0; ni < 4; ++ni)
        acc[mi][ni] = __builtin_amdgcn_wmma_f32_16x16x32_bf16(
            false, afrag[mi], false, bfrag[ni], (short)0, acc[mi][ni], false, false);
    __syncthreads();
  }

#pragma unroll
  for (int mi = 0; mi < 2; ++mi)
#pragma unroll
    for (int ni = 0; ni < 4; ++ni) {
      const int col = n0 + wn*64 + ni*16 + l16;
      float bv = bias0[col];
      if (HAS_B1) bv += bias1[col];
#pragma unroll
      for (int i = 0; i < 8; ++i) {
        const int row = m0 + wm*32 + mi*16 + half*8 + i;
        C[(size_t)row * N + col] = acc[mi][ni][i] + bv;
      }
    }
}

// ---------------------------------------------------------------------------
// Per-token top-64 of 16384 via LDS radix select. Row moves global<->LDS with
// async-to/from-LDS ops; select/mask/compact operate entirely in LDS.
// ---------------------------------------------------------------------------
__global__ __launch_bounds__(256) void topk_kernel(
    float* __restrict__ h, int* __restrict__ idxOut, float* __restrict__ valOut)
{
  __shared__ __align__(16) unsigned keys[D_HID];   // 64 KB row cache
  __shared__ unsigned hist[256];
  __shared__ unsigned sg[256], se[256];
  __shared__ unsigned sSel, sRem;

  const int n   = blockIdx.x;
  const int tid = threadIdx.x;
  constexpr int PER_T = D_HID / 256;   // 64 contiguous elements per thread

  float* row = h + (size_t)n * D_HID;
  const unsigned kBase = lds_addr(&keys[0]);

  // async copy row (raw f32 bits) into LDS: 4096 float4, 16 per thread
#pragma unroll
  for (int j = 0; j < 16; ++j) {
    const unsigned off = (unsigned)((tid + 256 * j) * 16);
    asm volatile("global_load_async_to_lds_b128 %0, %1, %2"
                 :: "v"(kBase + off), "v"(off), "s"((const float*)row) : "memory");
  }
  asm volatile("s_wait_asynccnt 0x0" ::: "memory");
  __syncthreads();

  // order-preserving key transform, in place
  for (int i = tid; i < D_HID; i += 256) {
    const unsigned u = keys[i];
    keys[i] = (u & 0x80000000u) ? ~u : (u | 0x80000000u);
  }
  __syncthreads();

  // 4x8-bit radix select for the k-th largest key
  unsigned prefix = 0, maskSoFar = 0, remaining = TOPK;
  for (int shift = 24; shift >= 0; shift -= 8) {
    hist[tid] = 0;
    __syncthreads();
    for (int j = 0; j < PER_T; ++j) {
      const unsigned key = keys[tid * PER_T + j];
      if ((key & maskSoFar) == prefix)
        atomicAdd(&hist[(key >> shift) & 255u], 1u);
    }
    __syncthreads();
    if (tid == 0) {
      unsigned cum = 0;
      for (int d = 255; d >= 0; --d) {
        const unsigned c = hist[d];
        if (cum + c >= remaining) { sSel = (unsigned)d; sRem = remaining - cum; break; }
        cum += c;
      }
    }
    __syncthreads();
    prefix    |= sSel << shift;
    maskSoFar |= 0xFFu << shift;
    remaining  = sRem;
    __syncthreads();
  }
  const unsigned T = prefix;     // key of the k-th largest element
  const unsigned r = remaining;  // ties (==T) to keep, first by index

  // deterministic rank among (>T) and (==T) in index order
  unsigned cg = 0, ce = 0;
  for (int j = 0; j < PER_T; ++j) {
    const unsigned key = keys[tid * PER_T + j];
    cg += (key > T); ce += (key == T);
  }
  sg[tid] = cg; se[tid] = ce;
  __syncthreads();
  for (int off = 1; off < 256; off <<= 1) {
    const unsigned vg = (tid >= off) ? sg[tid - off] : 0u;
    const unsigned ve = (tid >= off) ? se[tid - off] : 0u;
    __syncthreads();
    sg[tid] += vg; se[tid] += ve;
    __syncthreads();
  }
  const unsigned baseg = sg[tid] - cg;
  const unsigned basee = se[tid] - ce;
  const unsigned kGt   = TOPK - r;

  float* keysF = (float*)keys;
  unsigned rg = 0, re = 0;
  for (int j = 0; j < PER_T; ++j) {
    const int i = tid * PER_T + j;
    const unsigned key = keys[i];
    const unsigned u   = (key & 0x80000000u) ? (key & 0x7FFFFFFFu) : ~key;
    const float    f   = __uint_as_float(u);
    int slot = -1;
    if (key > T)       { slot = (int)(baseg + rg); ++rg; }
    else if (key == T) { if (basee + re < r) slot = (int)(kGt + basee + re); ++re; }
    float hv = 0.f;
    if (slot >= 0) {
      hv = fmaxf(f, 0.f);
      idxOut[n * TOPK + slot] = i;
      valOut[n * TOPK + slot] = hv;
    }
    keysF[i] = hv;  // only this thread touches element i in this phase
  }
  __syncthreads();  // drains DS stores; keysF fully built

  // async store h row LDS -> global
#pragma unroll
  for (int j = 0; j < 16; ++j) {
    const unsigned off = (unsigned)((tid + 256 * j) * 16);
    asm volatile("global_store_async_from_lds_b128 %0, %1, %2"
                 :: "v"(off), "v"(kBase + off), "s"(row) : "memory");
  }
  asm volatile("s_wait_asynccnt 0x0" ::: "memory");
}

// ---------------------------------------------------------------------------
// W_dec [D_OUT, D_HID] -> W_decT [D_HID, D_OUT] (then L2-resident: 128 MB)
// ---------------------------------------------------------------------------
__global__ __launch_bounds__(256) void transpose_kernel(
    const float* __restrict__ W, float* __restrict__ WT)
{
  __shared__ float tile[32][33];
  const int tx = threadIdx.x & 31;
  const int ty = threadIdx.x >> 5;
  const int bx = blockIdx.x;  // tile along D_HID
  const int by = blockIdx.y;  // tile along D_OUT
  const int x = bx*32 + tx;
  for (int j = ty; j < 32; j += 8)
    tile[j][tx] = W[(size_t)(by*32 + j) * D_HID + x];
  __syncthreads();
  const int xo = by*32 + tx;
  for (int j = ty; j < 32; j += 8)
    WT[(size_t)(bx*32 + j) * D_OUT + xo] = tile[tx][j];
}

// ---------------------------------------------------------------------------
// out[n,:] += sum_{s<64} val[n,s] * W_decT[idx[n,s], :]   (2.1 GFLOP total)
// ---------------------------------------------------------------------------
__global__ __launch_bounds__(256) void decode_kernel(
    const float* __restrict__ WT, const int* __restrict__ idx,
    const float* __restrict__ val, float* __restrict__ out)
{
  __shared__ int   sIdx[TOPK];
  __shared__ float sVal[TOPK];
  const int n   = blockIdx.x;
  const int tid = threadIdx.x;
  if (tid < TOPK) { sIdx[tid] = idx[n*TOPK + tid]; sVal[tid] = val[n*TOPK + tid]; }
  __syncthreads();
  float4 acc0 = {0,0,0,0}, acc1 = {0,0,0,0};
#pragma unroll 4
  for (int s = 0; s < TOPK; ++s) {
    const float4* rp = (const float4*)(WT + (size_t)sIdx[s] * D_OUT);
    const float v = sVal[s];
    const float4 a = rp[tid];
    const float4 b = rp[256 + tid];
    acc0.x += v*a.x; acc0.y += v*a.y; acc0.z += v*a.z; acc0.w += v*a.w;
    acc1.x += v*b.x; acc1.y += v*b.y; acc1.z += v*b.z; acc1.w += v*b.w;
  }
  float4* op = (float4*)(out + (size_t)n * D_OUT);
  float4 c0 = op[tid], c1 = op[256 + tid];
  c0.x += acc0.x; c0.y += acc0.y; c0.z += acc0.z; c0.w += acc0.w;
  c1.x += acc1.x; c1.y += acc1.y; c1.z += acc1.z; c1.w += acc1.w;
  op[tid] = c0; op[256 + tid] = c1;
}

// ---------------------------------------------------------------------------
// MSE loss: deterministic two-pass tree reduction.
// ---------------------------------------------------------------------------
__global__ __launch_bounds__(256) void loss_partial_kernel(
    const float* __restrict__ pred, const float* __restrict__ tgt,
    float* __restrict__ partial)
{
  __shared__ float red[256];
  const int total = NTOK * D_OUT;
  float s = 0.f;
  for (int i = blockIdx.x*256 + threadIdx.x; i < total; i += gridDim.x*256) {
    const float d = pred[i] - tgt[i];
    s += d * d;
  }
  red[threadIdx.x] = s; __syncthreads();
  for (int off = 128; off > 0; off >>= 1) {
    if (threadIdx.x < off) red[threadIdx.x] += red[threadIdx.x + off];
    __syncthreads();
  }
  if (threadIdx.x == 0) partial[blockIdx.x] = red[0];
}

__global__ __launch_bounds__(256) void loss_final_kernel(
    const float* __restrict__ partial, int nPart, float* __restrict__ lossOut)
{
  __shared__ float red[256];
  float s = 0.f;
  for (int i = threadIdx.x; i < nPart; i += 256) s += partial[i];
  red[threadIdx.x] = s; __syncthreads();
  for (int off = 128; off > 0; off >>= 1) {
    if (threadIdx.x < off) red[threadIdx.x] += red[threadIdx.x + off];
    __syncthreads();
  }
  if (threadIdx.x == 0)
    *lossOut = red[0] / (float)((size_t)NTOK * (size_t)D_OUT);
}

// ---------------------------------------------------------------------------
extern "C" void kernel_launch(void* const* d_in, const int* in_sizes, int n_in,
                              void* d_out, int out_size, void* d_ws, size_t ws_size,
                              hipStream_t stream)
{
  const float* x      = (const float*)d_in[0];
  const float* tgt    = (const float*)d_in[1];
  const float* W_enc  = (const float*)d_in[2];
  const float* b_enc  = (const float*)d_in[3];
  const float* W_dec  = (const float*)d_in[4];
  const float* b_dec  = (const float*)d_in[5];
  const float* W_skip = (const float*)d_in[6];
  const float* b_skip = (const float*)d_in[7];
  // d_in[8] = k (device scalar); problem constant TOPK=64 baked in.

  float* outHat = (float*)d_out;                         // [NTOK, D_OUT]
  float* hBuf   = (float*)d_out + (size_t)NTOK * D_OUT;  // [NTOK, D_HID]
  float* lossP  = hBuf + (size_t)NTOK * D_HID;           // [1]

  // workspace: W_decT (128 MB) | idx (2 MB) | val (2 MB) | loss partials
  char*  ws    = (char*)d_ws;
  float* WdecT = (float*)ws;
  int*   idxB  = (int*)  (ws + (size_t)D_HID * D_OUT * sizeof(float));
  float* valB  = (float*)((char*)idxB + (size_t)NTOK * TOPK * sizeof(int));
  float* part  = (float*)((char*)valB + (size_t)NTOK * TOPK * sizeof(float));
  constexpr int LOSS_BLOCKS = 1024;

  // 1) encode: pre_acts = x @ W_enc^T + b_enc  (into h buffer)
  gemm_nt_bias<false><<<(NTOK/BM) * (D_HID/BN), 256, 0, stream>>>(
      x, W_enc, b_enc, nullptr, hBuf, NTOK, D_HID, D_IN);
  // 2) transpose W_dec for row-major sparse gathers
  transpose_kernel<<<dim3(D_HID/32, D_OUT/32), 256, 0, stream>>>(W_dec, WdecT);
  // 3) per-token top-64 + ReLU (in place) + compacted (idx,val)
  topk_kernel<<<NTOK, 256, 0, stream>>>(hBuf, idxB, valB);
  // 4) skip path: out = x @ W_skip^T + b_dec + b_skip
  gemm_nt_bias<true><<<(NTOK/BM) * (D_OUT/BN), 256, 0, stream>>>(
      x, W_skip, b_dec, b_skip, outHat, NTOK, D_OUT, D_IN);
  // 5) sparse decode accumulate: out += h_sparse @ W_dec^T
  decode_kernel<<<NTOK, 256, 0, stream>>>(WdecT, idxB, valB, outHat);
  // 6-7) MSE loss
  loss_partial_kernel<<<LOSS_BLOCKS, 256, 0, stream>>>(outHat, tgt, part);
  loss_final_kernel<<<1, 256, 0, stream>>>(part, LOSS_BLOCKS, lossP);
}
